// VectorQuantizer_76836964925843
// MI455X (gfx1250) — matrix operand order repaired and checked by reading, not compiled, not run on hardware
//
#include <hip/hip_runtime.h>
#include <hip/hip_bf16.h>

// ---------------------------------------------------------------------------
// VQ-VAE forward for MI455X (gfx1250, wave32, WMMA + async-LDS DMA).
// Outputs (flat, in order): quantized_st[8*2048*512] f32, loss[1],
// encoding_indices[16384] (as float), z_e[8*2048*512] f32.
// ---------------------------------------------------------------------------

#define N_ROWS   16384   // 8*2048 flattened tokens
#define IN_DIM   1024
#define D_DIM    512
#define K_CODES  8192

typedef __attribute__((ext_vector_type(2)))  float  v2f;
typedef __attribute__((ext_vector_type(8)))  float  v8f;
typedef __attribute__((ext_vector_type(8)))  __bf16 v8bf;
typedef __attribute__((ext_vector_type(16))) __bf16 v16bf;

// ---------------------------------------------------------------------------
// CDNA5 async global->LDS copy (ASYNCcnt-tracked, see CDNA5 ISA).
// Inline asm is used because the clang builtin arity differs across
// toolchains. LDS address = low 32 bits of the flat shared pointer
// (aperture form, ISA 10.2).
// ---------------------------------------------------------------------------
__device__ __forceinline__ void async_copy16(const void* gsrc, void* ldst)
{
    const unsigned int loff = (unsigned int)(unsigned long long)ldst;
    const unsigned long long gaddr = (unsigned long long)gsrc;
    asm volatile("global_load_async_to_lds_b128 %0, %1, off"
                 :: "v"(loff), "v"(gaddr) : "memory");
}
__device__ __forceinline__ void wait_async0()
{
    asm volatile("s_wait_asynccnt 0x0" ::: "memory");
}

// ---------------------------------------------------------------------------
// Kernel 1: z_e = latents @ W + b   (f32 WMMA 16x16x4, exact fp32 path).
// One 16x16 output tile per wave; 8 waves per block.
// A frag (16x4 f32): lanes 0-15 hold M=lane, K={0,1}; lanes 16-31 M=lane-16,
// K={2,3}  -> one b64 load per lane per step.
// B frag (4x16 f32): lane holds column N=lane%16, K = 2*(lane/16)+{0,1}.
// C/D (16x16 f32): VGPR r -> row r + 8*(lane/16), col lane%16.
// ---------------------------------------------------------------------------
__global__ __launch_bounds__(256) void encode_gemm(
    const float* __restrict__ A,      // [N_ROWS, IN_DIM]
    const float* __restrict__ W,      // [IN_DIM, D_DIM]
    const float* __restrict__ bias,   // [D_DIM]
    float*  __restrict__ ze_out,      // [N_ROWS, D_DIM] (d_out z_e region)
    __bf16* __restrict__ zbf)         // [N_ROWS, D_DIM] bf16 copy (ws)
{
    const int lane = threadIdx.x & 31;
    const int wave = threadIdx.x >> 5;
    const int tile = blockIdx.x * 8 + wave;     // 1024*32 tiles total
    const int mt = tile >> 5;                    // M tile (of 1024)
    const int nt = tile & 31;                    // N tile (of 32)
    const int m0 = mt * 16, n0 = nt * 16;
    const int col = lane & 15;
    const int hf  = lane >> 4;                   // K-half selector

    const float bv = bias[n0 + col];
    v8f c;
#pragma unroll
    for (int r = 0; r < 8; ++r) c[r] = bv;       // fold bias into accumulator

    const float* arow = A + (size_t)(m0 + col) * IN_DIM;

#pragma unroll 4
    for (int kk = 0; kk < IN_DIM; kk += 4) {
        const int ka = kk + 2 * hf;
        v2f a, b;
        a.x = arow[ka];
        a.y = arow[ka + 1];
        b.x = W[(size_t)ka * D_DIM + n0 + col];
        b.y = W[(size_t)(ka + 1) * D_DIM + n0 + col];
        c = __builtin_amdgcn_wmma_f32_16x16x4_f32(
                false, a, false, b, (short)0, c, false, false);
    }

#pragma unroll
    for (int r = 0; r < 8; ++r) {
        const int row = m0 + r + 8 * hf;
        const size_t off = (size_t)row * D_DIM + n0 + col;
        ze_out[off] = c[r];
        zbf[off]    = (__bf16)c[r];
    }
}

// ---------------------------------------------------------------------------
// Kernel 2: codebook -> bf16, and cnorm[k] = ||E_k||^2 (fp32). Warp per row.
// ---------------------------------------------------------------------------
__global__ __launch_bounds__(256) void prep_codebook(
    const float* __restrict__ cb, __bf16* __restrict__ cbf,
    float* __restrict__ cnorm)
{
    const int lane = threadIdx.x & 31;
    const int warp = threadIdx.x >> 5;
    const int row  = blockIdx.x * 8 + warp;       // 1024 blocks -> 8192 rows
    const float* src = cb + (size_t)row * D_DIM;
    __bf16*      dst = cbf + (size_t)row * D_DIM;
    float s = 0.f;
#pragma unroll
    for (int i = 0; i < D_DIM / 32; ++i) {
        const float v = src[lane + i * 32];
        dst[lane + i * 32] = (__bf16)v;
        s += v * v;
    }
#pragma unroll
    for (int off = 16; off; off >>= 1) s += __shfl_down(s, off, 32);
    if (lane == 0) cnorm[row] = s;
}

// ---------------------------------------------------------------------------
// Kernel 3: argmin_k ( -2 * Z @ E^T + ||E||^2 )   (bf16 WMMA 16x16x32).
// Block = 8 waves = 128 rows. Z block async-DMA'd into LDS once; the 16-code
// B tile is async-DMA'd into a double-buffered LDS slot each iteration
// (issue nt+1, compute nt, s_wait_asynccnt 0 + barrier).
// Rows padded to 520 bf16 to break the 1024B-stride bank pattern.
// Each wave owns one 16-row tile, sweeps all 512 code tiles keeping a
// per-lane running (min, argmin), then reduces across lanes with shfl_xor
// (index tie-break == jnp.argmin first-min).
// A frag (bf16 16x32): lane m=lane%16; K = 0..15 (lanes<16) or
// 16..31 -> two contiguous 16B LDS reads.
// B frag (bf16 32x16): lane owns column n=lane%16, K = 16*(lane/16)+0..15 ->
// 32 contiguous bytes of a codebook row (row-major bf16, no transpose).
// ---------------------------------------------------------------------------
#define ZSH_STRIDE 520                 // 512 + 8 bf16 pad
#define ZSH_ELEMS  (128 * ZSH_STRIDE)
#define BSH_ELEMS  (16 * ZSH_STRIDE)
#define ARGMIN_LDS ((ZSH_ELEMS + 2 * BSH_ELEMS) * 2)   // bytes = 166,400

__device__ __forceinline__ void stage_btile(
    const __bf16* __restrict__ cbf, int n0, __bf16* bbuf, int tid)
{
#pragma unroll
    for (int c = tid; c < 16 * 64; c += 256) {        // 1024 x 16B chunks
        const int row = c >> 6, c8 = c & 63;
        async_copy16(cbf + (size_t)(n0 + row) * D_DIM + c8 * 8,
                     bbuf + row * ZSH_STRIDE + c8 * 8);
    }
}

__global__ __launch_bounds__(256) void vq_argmin(
    const __bf16* __restrict__ zbf, const __bf16* __restrict__ cbf,
    const float* __restrict__ cnorm,
    int* __restrict__ amin, float* __restrict__ idx_out)
{
    extern __shared__ __bf16 lds[];
    __bf16* zsh   = lds;                               // [128][ZSH_STRIDE]
    __bf16* bbuf0 = lds + ZSH_ELEMS;                   // [16][ZSH_STRIDE]
    __bf16* bbuf1 = bbuf0 + BSH_ELEMS;                 // [16][ZSH_STRIDE]
    const int tid      = threadIdx.x;
    const int rowblock = blockIdx.x * 128;

    // async stage: 128 rows x 512 bf16 of Z, plus first B tile
    for (int c = tid; c < 128 * 64; c += 256) {        // 8192 x 16B chunks
        const int row = c >> 6, c8 = c & 63;
        async_copy16(zbf + (size_t)(rowblock + row) * D_DIM + c8 * 8,
                     &zsh[row * ZSH_STRIDE + c8 * 8]);
    }
    stage_btile(cbf, 0, bbuf0, tid);
    wait_async0();
    __syncthreads();

    const int lane = tid & 31;
    const int wave = tid >> 5;
    const int col  = lane & 15;
    const int hf   = lane >> 4;
    const __bf16* myrow = &zsh[(wave * 16 + col) * ZSH_STRIDE];

    float dmin[8];
    int   darg[8];
#pragma unroll
    for (int r = 0; r < 8; ++r) { dmin[r] = 3.0e38f; darg[r] = 0; }

    for (int nt = 0; nt < K_CODES / 16; ++nt) {
        const __bf16* bcur = (nt & 1) ? bbuf1 : bbuf0;
        if (nt + 1 < K_CODES / 16)                      // prefetch next tile
            stage_btile(cbf, (nt + 1) * 16, (nt & 1) ? bbuf0 : bbuf1, tid);

        const int n0 = nt * 16;
        const float cn = cnorm[n0 + col];
        const __bf16* brow = bcur + col * ZSH_STRIDE + hf * 16;

        v8f s = {};
#pragma unroll
        for (int ks = 0; ks < 16; ++ks) {
            const int k0 = ks * 32;
            v8bf alo = *reinterpret_cast<const v8bf*>(myrow + k0 + hf * 16);
            v8bf ahi = *reinterpret_cast<const v8bf*>(myrow + k0 + hf * 16 + 8);
            v16bf a;
            reinterpret_cast<v8bf*>(&a)[0] = alo;
            reinterpret_cast<v8bf*>(&a)[1] = ahi;
            const v16bf b = *reinterpret_cast<const v16bf*>(brow + k0);
            s = __builtin_amdgcn_wmma_f32_16x16x32_bf16(
                    false, a, false, b, (short)0, s, false, false);
        }
#pragma unroll
        for (int r = 0; r < 8; ++r) {
            const float d = cn - 2.0f * s[r];     // + ||z||^2 is row-constant
            if (d < dmin[r]) { dmin[r] = d; darg[r] = n0 + col; }
        }

        wait_async0();      // next B tile resident before anyone touches it
        __syncthreads();    // and all waves done with the current buffers
    }

    // reduce across the 16 lanes of each half-wave (rows r+8*hf)
#pragma unroll
    for (int r = 0; r < 8; ++r) {
        float v = dmin[r];
        int   id = darg[r];
        for (int off = 8; off; off >>= 1) {
            const float ov = __shfl_xor(v, off, 16);
            const int   oi = __shfl_xor(id, off, 16);
            if (ov < v || (ov == v && oi < id)) { v = ov; id = oi; }
        }
        if (col == 0) {
            const int row = rowblock + wave * 16 + r + 8 * hf;
            amin[row]    = id;
            idx_out[row] = (float)id;
        }
    }
}

// ---------------------------------------------------------------------------
// Kernel 4/5/6: gather + loss = 1.25 * mean((q - z_e)^2); write quantized.
// ---------------------------------------------------------------------------
__global__ void zero_acc(float* acc) { acc[0] = 0.f; }

__global__ __launch_bounds__(256) void gather_loss(
    const float* __restrict__ cb, const int* __restrict__ amin,
    const float* __restrict__ ze, float* __restrict__ qout,
    float* __restrict__ acc)
{
    __shared__ float red[8];
    const int lane = threadIdx.x & 31;
    const int warp = threadIdx.x >> 5;
    const int row  = blockIdx.x * 8 + warp;       // 2048 blocks -> 16384 rows
    const float* crow = cb + (size_t)amin[row] * D_DIM;
    const float* zrow = ze + (size_t)row * D_DIM;
    float*       q    = qout + (size_t)row * D_DIM;
    float s = 0.f;
#pragma unroll
    for (int i = 0; i < D_DIM / 32; ++i) {
        const int e = lane + i * 32;
        const float qv = crow[e];
        q[e] = qv;                                 // quantized_st == quantized
        const float d = qv - zrow[e];
        s += d * d;
    }
#pragma unroll
    for (int off = 16; off; off >>= 1) s += __shfl_down(s, off, 32);
    if (lane == 0) red[warp] = s;
    __syncthreads();
    if (threadIdx.x == 0) {
        float t = 0.f;
        for (int w = 0; w < 8; ++w) t += red[w];
        atomicAdd(acc, t);
    }
}

__global__ void finalize_loss(const float* acc, float* loss)
{
    loss[0] = 1.25f * acc[0] / (float)((double)N_ROWS * (double)D_DIM);
}

// ---------------------------------------------------------------------------
extern "C" void kernel_launch(void* const* d_in, const int* in_sizes, int n_in,
                              void* d_out, int out_size, void* d_ws, size_t ws_size,
                              hipStream_t stream)
{
    const float* latents  = (const float*)d_in[0];   // [8,2048,1024]
    const float* W        = (const float*)d_in[1];   // [1024,512]
    const float* b        = (const float*)d_in[2];   // [512]
    const float* codebook = (const float*)d_in[3];   // [8192,512]

    float* out  = (float*)d_out;
    float* q    = out;                               // quantized_st
    float* loss = out + (size_t)N_ROWS * D_DIM;      // 8388608
    float* idxf = loss + 1;                          // 8388609
    float* ze   = idxf + N_ROWS;                     // 8404993

    char* ws = (char*)d_ws;
    __bf16* zbf   = (__bf16*)ws;                                   // 16 MB
    __bf16* cbf   = (__bf16*)(ws + (size_t)N_ROWS * D_DIM * 2);    //  8 MB
    float*  cnorm = (float*)(ws + (size_t)N_ROWS * D_DIM * 2
                                + (size_t)K_CODES * D_DIM * 2);    // 32 KB
    int*    amin  = (int*)((char*)cnorm + K_CODES * sizeof(float));
    float*  acc   = (float*)((char*)amin + N_ROWS * sizeof(int));

    encode_gemm  <<<(N_ROWS / 16) * (D_DIM / 16) / 8, 256, 0, stream>>>(
        latents, W, b, ze, zbf);
    prep_codebook<<<K_CODES / 8, 256, 0, stream>>>(codebook, cbf, cnorm);
    zero_acc     <<<1, 1, 0, stream>>>(acc);
    vq_argmin    <<<N_ROWS / 128, 256, ARGMIN_LDS, stream>>>(
        zbf, cbf, cnorm, amin, idxf);
    gather_loss  <<<N_ROWS / 8, 256, 0, stream>>>(codebook, amin, ze, q, acc);
    finalize_loss<<<1, 1, 0, stream>>>(acc, loss);
}
